// UnaGenModel_35613868818612
// MI455X (gfx1250) — compile-verified, matrix operand order checked by reading, not run on hardware
//
#include <hip/hip_runtime.h>
#include <hip/hip_fp16.h>

// ---------------------------------------------------------------------------
// CDNA5 (gfx1250, wave32) implementation of the UnaGen reference:
//   occ = MLP(interp(UNet_geo(img), coo));  rgb = MLP(interp(UNet_ren(img), coo))
// All conv / conv-transpose / 1x1 / MLP layers run through
// v_wmma_f32_16x16x32_f16 (fp32 accumulate). Weight tiles staged to LDS as f16
// (async global->LDS DMA where available), fragments read back as ds_load_b128
// vectors, branch-free mask-multiplied B gathers (loads stay straight-line so
// the scheduler can cluster them), 2 D tiles per wave per K-chunk.
// ---------------------------------------------------------------------------

typedef __attribute__((ext_vector_type(16))) _Float16 v16h;
typedef __attribute__((ext_vector_type(8)))  _Float16 v8h;
typedef __attribute__((ext_vector_type(8)))  float    v8f;
typedef __attribute__((ext_vector_type(4)))  float    f4;

#if defined(__has_builtin)
#if __has_builtin(__builtin_amdgcn_global_load_async_to_lds_b32) && \
    __has_builtin(__builtin_amdgcn_s_wait_asynccnt)
#define HAVE_ASYNC_LDS 1
#endif
#endif
#ifndef HAVE_ASYNC_LDS
#define HAVE_ASYNC_LDS 0
#endif

// K index inside a 32-wide chunk for fragment half h (0..15), lane-group g:
// halfs 0..7 -> K = g*8 + h ; halfs 8..15 -> K = 16 + g*8 + (h-8)
__device__ __forceinline__ int wmma_kpat(int h, int g) {
    return ((h & 8) << 1) + (g << 3) + (h & 7);
}

// load one 16-half fragment from LDS (two aligned 16-byte ds_load_b128)
__device__ __forceinline__ v16h frag_lds(const _Float16* p /* = &wl[row*Kpad + k0] */, int g) {
    v8h lo = *(const v8h*)(p + g * 8);
    v8h hi = *(const v8h*)(p + 16 + g * 8);
    return __builtin_shufflevector(lo, hi, 0,1,2,3,4,5,6,7,8,9,10,11,12,13,14,15);
}

static inline int ceil_div_i(int a, int b) { return (a + b - 1) / b; }

// ---------------------------------------------------------------------------
// 3x3 conv, pad=1 (SAME), NCHW fp32 in/out, f16 WMMA compute, optional ReLU.
// Implicit GEMM, tap-major im2col: K = 9*Cp, k = tap*Cp + ci, Cp = 2^cshift.
// Each 32-chunk lies inside one tap -> tap/dy/dx uniform SALU. B gathers load
// a clamped (always-valid) address unconditionally and multiply by a 0/1 mask,
// so no exec branches and no per-load waits in the inner loop.
// Block = 128 threads = 4 waves; wave owns 16 couts x 32 pixels (2 D tiles).
// ---------------------------------------------------------------------------
__global__ void __launch_bounds__(128)
conv3x3_wmma(const float* __restrict__ in, const float* __restrict__ wgt,
             const float* __restrict__ bias, float* __restrict__ out,
             int B, int Cin, int Cout, int H, int W, int relu, int cshift)
{
    __shared__ _Float16 wl[16 * 1152];           // 16 x (9*Cp), Cp<=128
    const int Cp   = 1 << cshift;
    const int Kp   = 9 * Cp;
    const int co0  = blockIdx.y * 16;

    // weights: wl[m][t*Cp + ci] = wgt[(co0+m)*Cin*9 + ci*9 + t] (zero padded)
    for (int idx = threadIdx.x; idx < 16 * Kp; idx += blockDim.x) {
        int m  = idx / Kp, r = idx - m * Kp;
        int t  = r >> cshift;
        int ci = r & (Cp - 1);
        int co = co0 + m;
        float wv = 0.f;
        if (co < Cout && ci < Cin) wv = wgt[(co * Cin + ci) * 9 + t];
        wl[m * Kp + r] = (_Float16)wv;
    }
    __syncthreads();

    const int lane = threadIdx.x & 31;
    const int wave = threadIdx.x >> 5;
    const int g    = lane >> 4;
    const int ln   = lane & 15;
    const int HW   = H * W;
    const int npix = B * HW;
    const int pix0 = (blockIdx.x * 4 + wave) * 32;

    // two pixel subtiles per wave
    int pixS[2] = {pix0 + ln, pix0 + 16 + ln};
    int pbS[2], pyS[2], pxS[2];
    bool pvS[2];
#pragma unroll
    for (int s = 0; s < 2; ++s) {
        pvS[s] = pixS[s] < npix;
        int p = pvS[s] ? pixS[s] : 0;
        pbS[s] = p / HW;
        int r = p - pbS[s] * HW;
        pyS[s] = r / W;
        pxS[s] = r - pyS[s] * W;
    }

    const bool ci_full = (Cin == Cp);
    // per-half gather offsets (loop-invariant): pat*HW, with ragged-ci fold
    int  hwof[16];
    bool hval[16];
#pragma unroll
    for (int h = 0; h < 16; ++h) {
        int pat = wmma_kpat(h, g);
        hval[h] = ci_full || (pat < Cin);
        hwof[h] = (hval[h] ? pat : 0) * HW;
    }

    v8f acc0 = {}, acc1 = {};

    for (int k0 = 0; k0 < Kp; k0 += 32) {
        const int t   = k0 >> cshift;            // uniform tap 0..8
        const int ci0 = k0 & (Cp - 1);
        const int dy  = t / 3 - 1;               // uniform SALU
        const int dx  = t - (t / 3) * 3 - 1;

        v16h a = frag_lds(&wl[ln * Kp + k0], g); // 2x ds_load_b128

        v16h bb[2];
#pragma unroll
        for (int s = 0; s < 2; ++s) {
            const int yy = pyS[s] + dy, xx = pxS[s] + dx;
            const bool ok = pvS[s] && yy >= 0 && yy < H && xx >= 0 && xx < W;
            const float msk = ok ? 1.f : 0.f;     // 0/1 multiply mask
            const int yyc = yy < 0 ? 0 : (yy >= H ? H - 1 : yy);
            const int xxc = xx < 0 ? 0 : (xx >= W ? W - 1 : xx);
            const int off = (pbS[s] * Cin + ci0) * HW + yyc * W + xxc;   // 32-bit
            const float* base = in + off;
            __builtin_prefetch(base + 8 * (size_t)HW);                    // speculative
#pragma unroll
            for (int h = 0; h < 16; ++h) {
                float v = base[hwof[h]];          // always in-bounds (clamped),
                                                  // result used unconditionally
                bb[s][h] = (_Float16)(v * (hval[h] ? msk : 0.f));
            }
        }
        acc0 = __builtin_amdgcn_wmma_f32_16x16x32_f16(false, a, false, bb[0],
                                                      (short)0, acc0, false, false);
        acc1 = __builtin_amdgcn_wmma_f32_16x16x32_f16(false, a, false, bb[1],
                                                      (short)0, acc1, false, false);
    }

#pragma unroll
    for (int s = 0; s < 2; ++s) {
        const v8f& acc = s ? acc1 : acc0;
        if (pvS[s]) {
#pragma unroll
            for (int r = 0; r < 8; ++r) {
                int co = co0 + r + g * 8;         // D: M = vgpr + 8*(lane>=16)
                if (co < Cout) {
                    float o = acc[r] + bias[co];
                    if (relu) o = fmaxf(o, 0.f);
                    out[((pbS[s] * Cout + co) * H + pyS[s]) * W + pxS[s]] = o;
                }
            }
        }
    }
}

// ---------------------------------------------------------------------------
// Per-pixel GEMM: out[b, co_base+co, y*sy+oy, x*sx+ox] =
//   sum_ci in[b,ci,y,x] * wgt[(co*Cin+ci)*wstride + woff] + bias[co]
// Covers conv_transpose 2x2/s2 taps (wstride=4) and the 1x1 conv (wstride=1).
// Weight tile staged via async global->LDS DMA when available (ASYNCcnt).
// ---------------------------------------------------------------------------
__global__ void __launch_bounds__(128)
gemm_pix_wmma(const float* __restrict__ in, const float* __restrict__ wgt,
              const float* __restrict__ bias, float* __restrict__ out,
              int B, int Cin, int Cout, int H, int W,
              int wstride, int woff, int sy, int oy, int sx, int ox,
              int Ctot, int co_base, int relu, int cshift)
{
    __shared__ _Float16 wl[16 * 128];            // 16 x Cp, Cp<=128
    const int Cp  = 1 << cshift;
    const int co0 = blockIdx.y * 16;

#if HAVE_ASYNC_LDS
    __shared__ float wraw[16 * 128];
    for (int idx = threadIdx.x; idx < 16 * Cp; idx += blockDim.x) {
        int m = idx / Cp, ci = idx - m * Cp;
        int co = co0 + m;
        if (co < Cout && ci < Cin)
            __builtin_amdgcn_global_load_async_to_lds_b32(
                (int*)(wgt + (size_t)(co * Cin + ci) * wstride + woff),
                (int*)(wraw + idx), 0, 0);
    }
    __builtin_amdgcn_s_wait_asynccnt(0);
    __syncthreads();
    for (int idx = threadIdx.x; idx < 16 * Cp; idx += blockDim.x) {
        int m = idx / Cp, ci = idx - m * Cp;
        int co = co0 + m;
        wl[idx] = (co < Cout && ci < Cin) ? (_Float16)wraw[idx] : (_Float16)0.f;
    }
#else
    for (int idx = threadIdx.x; idx < 16 * Cp; idx += blockDim.x) {
        int m = idx / Cp, ci = idx - m * Cp;
        int co = co0 + m;
        float wv = 0.f;
        if (co < Cout && ci < Cin) wv = wgt[(co * Cin + ci) * wstride + woff];
        wl[idx] = (_Float16)wv;
    }
#endif
    __syncthreads();

    const int lane = threadIdx.x & 31;
    const int wave = threadIdx.x >> 5;
    const int g    = lane >> 4;
    const int ln   = lane & 15;
    const int HW   = H * W;
    const int npix = B * HW;
    const int pix0 = (blockIdx.x * 4 + wave) * 32;

    int pixS[2] = {pix0 + ln, pix0 + 16 + ln};
    int pbS[2], pyS[2], pxS[2], offS[2];
    bool pvS[2];
#pragma unroll
    for (int s = 0; s < 2; ++s) {
        pvS[s] = pixS[s] < npix;
        int p = pvS[s] ? pixS[s] : 0;
        pbS[s] = p / HW;
        int r = p - pbS[s] * HW;
        pyS[s] = r / W;
        pxS[s] = r - pyS[s] * W;
        offS[s] = pbS[s] * Cin * HW + pyS[s] * W + pxS[s];   // 32-bit base
    }

    const bool ci_full = (Cin == Cp);
    int  hwof[16];
    bool hval[16];
#pragma unroll
    for (int h = 0; h < 16; ++h) {
        int pat = wmma_kpat(h, g);
        hval[h] = ci_full || (pat < Cin);     // k0==0 whenever Cin < Cp
        hwof[h] = (hval[h] ? pat : 0) * HW;
    }

    v8f acc0 = {}, acc1 = {};

    for (int k0 = 0; k0 < Cp; k0 += 32) {
        v16h a = frag_lds(&wl[ln * Cp + k0], g);
        v16h bb[2];
#pragma unroll
        for (int s = 0; s < 2; ++s) {
            const float msk = pvS[s] ? 1.f : 0.f;
            const float* base = in + offS[s] + k0 * HW;
#pragma unroll
            for (int h = 0; h < 16; ++h) {
                float v = base[hwof[h]];
                bb[s][h] = (_Float16)(v * (hval[h] ? msk : 0.f));
            }
        }
        acc0 = __builtin_amdgcn_wmma_f32_16x16x32_f16(false, a, false, bb[0],
                                                      (short)0, acc0, false, false);
        acc1 = __builtin_amdgcn_wmma_f32_16x16x32_f16(false, a, false, bb[1],
                                                      (short)0, acc1, false, false);
    }

    const int Ho = H * sy, Wo = W * sx;
#pragma unroll
    for (int s = 0; s < 2; ++s) {
        const v8f& acc = s ? acc1 : acc0;
        if (pvS[s]) {
            const int yo = pyS[s] * sy + oy, xo = pxS[s] * sx + ox;
#pragma unroll
            for (int r = 0; r < 8; ++r) {
                int co = co0 + r + g * 8;
                if (co < Cout) {
                    float o = acc[r] + bias[co];
                    if (relu) o = fmaxf(o, 0.f);
                    out[((pbS[s] * Ctot + co_base + co) * Ho + yo) * Wo + xo] = o;
                }
            }
        }
    }
}

// ---------------------------------------------------------------------------
// MLP layer: out[p,n] = act( sum_k x[p*160+k]*wgt[n*K+k] + bias[n] )
// Row stride fixed at 160 (= max Kpad); producers zero-fill cols [Nout,160)
// so A-side float4 vector loads never need a k<K guard.
// Wave owns 32 rows x 16 cols (2 D tiles). act: 0 relu, 1 sigmoid.
// padTo: zero-fill output cols [Nout, padTo) (0 for the final layer).
// ---------------------------------------------------------------------------
#define MLP_LD 160

__global__ void __launch_bounds__(128)
mlp_wmma(const float* __restrict__ x, const float* __restrict__ wgt,
         const float* __restrict__ bias, float* __restrict__ out,
         int P, int K, int Nout, int ldo, int act, int padTo)
{
    __shared__ _Float16 wl[16 * MLP_LD];
    const int Kpad = (K + 31) & ~31;
    const int n0   = blockIdx.y * 16;

#if HAVE_ASYNC_LDS
    __shared__ float wraw[16 * MLP_LD];
    for (int idx = threadIdx.x; idx < 16 * Kpad; idx += blockDim.x) {
        int n = idx / Kpad, k = idx - n * Kpad;
        if (n0 + n < Nout && k < K)
            __builtin_amdgcn_global_load_async_to_lds_b32(
                (int*)(wgt + (size_t)(n0 + n) * K + k),
                (int*)(wraw + idx), 0, 0);
    }
    __builtin_amdgcn_s_wait_asynccnt(0);
    __syncthreads();
    for (int idx = threadIdx.x; idx < 16 * Kpad; idx += blockDim.x) {
        int n = idx / Kpad, k = idx - n * Kpad;
        wl[idx] = (n0 + n < Nout && k < K) ? (_Float16)wraw[idx] : (_Float16)0.f;
    }
#else
    for (int idx = threadIdx.x; idx < 16 * Kpad; idx += blockDim.x) {
        int n = idx / Kpad, k = idx - n * Kpad;
        float wv = 0.f;
        if (n0 + n < Nout && k < K) wv = wgt[(n0 + n) * K + k];
        wl[idx] = (_Float16)wv;
    }
#endif
    __syncthreads();

    const int lane = threadIdx.x & 31;
    const int wave = threadIdx.x >> 5;
    const int g    = lane >> 4;
    const int ln   = lane & 15;
    const int row0 = (blockIdx.x * 4 + wave) * 32;    // P assumed multiple of 128
    const int rowS[2] = {row0 + ln, row0 + 16 + ln};

    v8f acc0 = {}, acc1 = {};
    for (int k0 = 0; k0 < Kpad; k0 += 32) {
        v16h bfrag = frag_lds(&wl[ln * Kpad + k0], g);   // B: N = ln
        v16h aS[2];
#pragma unroll
        for (int s = 0; s < 2; ++s) {
            const float* xp = x + (size_t)rowS[s] * MLP_LD + k0 + g * 8;
            f4 lo0 = *(const f4*)(xp);
            f4 lo1 = *(const f4*)(xp + 4);
            f4 hi0 = *(const f4*)(xp + 16);
            f4 hi1 = *(const f4*)(xp + 20);
#pragma unroll
            for (int h = 0; h < 4; ++h) {
                aS[s][h]      = (_Float16)lo0[h];
                aS[s][h + 4]  = (_Float16)lo1[h];
                aS[s][h + 8]  = (_Float16)hi0[h];
                aS[s][h + 12] = (_Float16)hi1[h];
            }
        }
        acc0 = __builtin_amdgcn_wmma_f32_16x16x32_f16(false, aS[0], false, bfrag,
                                                      (short)0, acc0, false, false);
        acc1 = __builtin_amdgcn_wmma_f32_16x16x32_f16(false, aS[1], false, bfrag,
                                                      (short)0, acc1, false, false);
    }

    const int col = n0 + ln;
    if (col < padTo || col < Nout) {
        float bv = (col < Nout) ? bias[col] : 0.f;
#pragma unroll
        for (int s = 0; s < 2; ++s) {
            const v8f& acc = s ? acc1 : acc0;
            int rbase = row0 + (s ? 16 : 0);
#pragma unroll
            for (int r = 0; r < 8; ++r) {
                int rr = rbase + r + g * 8;
                float o = 0.f;
                if (col < Nout) {
                    o = acc[r] + bv;
                    if (act == 0)      o = fmaxf(o, 0.f);
                    else if (act == 1) o = 1.f / (1.f + __expf(-o));
                }
                out[(size_t)rr * ldo + col] = o;
            }
        }
    }
}

// ---------------------------------------------------------------------------
// Elementwise helpers
// ---------------------------------------------------------------------------
__global__ void maxpool2_kernel(const float* __restrict__ in, float* __restrict__ out,
                                int BC, int H, int W)
{
    const int Ho = H >> 1, Wo = W >> 1;
    int idx = blockIdx.x * blockDim.x + threadIdx.x;
    int tot = BC * Ho * Wo;
    if (idx >= tot) return;
    int c = idx / (Ho * Wo);
    int r = idx - c * (Ho * Wo);
    int y = r / Wo, xx = r - y * Wo;
    const float* p = in + ((size_t)c * H + 2 * y) * W + 2 * xx;
    out[idx] = fmaxf(fmaxf(p[0], p[1]), fmaxf(p[W], p[W + 1]));
}

__global__ void copy_into_cat(const float* __restrict__ src, float* __restrict__ dst,
                              int B, int C, int HW, int Ctot, int co_base)
{
    int idx = blockIdx.x * blockDim.x + threadIdx.x;
    int tot = B * C * HW;
    if (idx >= tot) return;
    int b = idx / (C * HW);
    int r = idx - b * (C * HW);
    int c = r / HW;
    int p = r - c * HW;
    dst[((size_t)(b * Ctot + co_base + c)) * HW + p] = src[idx];
}

// order-preserving float <-> uint encoding for atomic min/max
__device__ __forceinline__ unsigned fenc(float f) {
    unsigned b = __float_as_uint(f);
    return (b & 0x80000000u) ? ~b : (b | 0x80000000u);
}
__device__ __forceinline__ float fdec(unsigned u) {
    unsigned b = (u & 0x80000000u) ? (u & 0x7FFFFFFFu) : ~u;
    return __uint_as_float(b);
}

__global__ void minmax_init(unsigned* stats) {
    if (threadIdx.x < 2) {
        stats[threadIdx.x * 2 + 0] = 0xFFFFFFFFu;  // min slot
        stats[threadIdx.x * 2 + 1] = 0u;           // max slot
    }
}

// stats = [ymin, ymax, xmin, xmax] (encoded)
__global__ void minmax_coo(const float* __restrict__ coo, int npts, unsigned* stats) {
    int i = blockIdx.x * blockDim.x + threadIdx.x;
    if (i >= npts) return;
    float y = coo[i * 3 + 0], x = coo[i * 3 + 1];
    atomicMin(&stats[0], fenc(y)); atomicMax(&stats[1], fenc(y));
    atomicMin(&stats[2], fenc(x)); atomicMax(&stats[3], fenc(x));
}

// gather feats[b,:,fy,fx] (C ch) + 3 coords into a (npts x LD) row matrix,
// zero-padding cols [C+3, LD)
__global__ void interp_gather(const float* __restrict__ feats, const float* __restrict__ coo,
                              const unsigned* __restrict__ stats, float* __restrict__ xrow,
                              int npts, int N, int C, int H, int W, int LD)
{
    long long idx = (long long)blockIdx.x * blockDim.x + threadIdx.x;
    long long tot = (long long)npts * LD;
    if (idx >= tot) return;
    int i = (int)(idx / LD);
    int c = (int)(idx - (long long)i * LD);
    float o = 0.f;
    if (c < C) {
        float ymin = fdec(stats[0]), ymax = fdec(stats[1]);
        float xmin = fdec(stats[2]), xmax = fdec(stats[3]);
        float y = coo[i * 3 + 0], x = coo[i * 3 + 1];
        int fy = (int)((y - ymin) / (ymax - ymin) * (float)(H - 1));
        int fx = (int)((x - xmin) / (xmax - xmin) * (float)(W - 1));
        fy = fy < 0 ? 0 : (fy > H - 1 ? H - 1 : fy);
        fx = fx < 0 ? 0 : (fx > W - 1 ? W - 1 : fx);
        int b = i / N;
        o = feats[((size_t)(b * C + c) * H + fy) * W + fx];
    } else if (c < C + 3) {
        o = coo[i * 3 + (c - C)];
    }
    xrow[(size_t)i * LD + c] = o;
}

// ---------------------------------------------------------------------------
// Host-side orchestration
// Input pytree order (setup_inputs dict order; nested dicts alphabetical):
//   0: image  1: voxels_coo
//   2..37 geo_params, 38..73 ren_params  (per UNet: bott[4], enc[12], out[2], ups[18])
//   74..83 occ_params, 84..93 rgb_params (5 x (w,b))
// ---------------------------------------------------------------------------
static inline int pow2_shift_ge32(int c) {
    int s = 5;                 // Cp = 32
    while ((1 << s) < c) ++s;  // 3,16,32 -> 5; 64 -> 6; 128 -> 7
    return s;
}

extern "C" void kernel_launch(void* const* d_in, const int* in_sizes, int n_in,
                              void* d_out, int out_size, void* d_ws, size_t ws_size,
                              hipStream_t stream)
{
    (void)in_sizes; (void)n_in; (void)out_size; (void)ws_size;

    const float* image = (const float*)d_in[0];
    const float* coo   = (const float*)d_in[1];
    float*       outp  = (float*)d_out;

    auto Pp = [&](int i) { return (const float*)d_in[i]; };

    // ---- workspace layout (fp32 elems; ~224 MB total) ----
    float*    wsf   = (float*)d_ws;
    unsigned* stats = (unsigned*)wsf;       // 4 encoded uints
    size_t off = 16;
    auto alloc = [&](size_t n) { float* p = wsf + off; off += n; return p; };
    float* tmp  = alloc(2u * 16 * 256 * 256);   // double-conv intermediate (max)
    float* e1   = alloc(2u * 16 * 256 * 256);
    float* e1p  = alloc(2u * 16 * 128 * 128);
    float* e2   = alloc(2u * 32 * 128 * 128);
    float* e2p  = alloc(2u * 32 * 64 * 64);
    float* e3   = alloc(2u * 64 * 64 * 64);
    float* e3p  = alloc(2u * 64 * 32 * 32);
    float* bA   = alloc(2u * 128 * 32 * 32);
    float* bB   = alloc(2u * 128 * 32 * 32);
    float* cat3 = alloc(2u * 128 * 64 * 64);
    float* d3   = alloc(2u * 64 * 64 * 64);
    float* cat2 = alloc(2u * 64 * 128 * 128);
    float* d2   = alloc(2u * 32 * 128 * 128);
    float* cat1 = alloc(2u * 32 * 256 * 256);
    float* d1   = alloc(2u * 16 * 256 * 256);
    float* feat = alloc(2u * 128 * 256 * 256);
    float* mlpA = alloc(65536u * MLP_LD);
    float* mlpB = alloc(65536u * MLP_LD);

    const int NPTS = 65536;   // B*N = 2*32^3

    auto conv3 = [&](const float* in, const float* w, const float* b, float* out,
                     int Ci, int Co, int H, int W, int relu) {
        dim3 grid(ceil_div_i(2 * H * W, 128), ceil_div_i(Co, 16));
        conv3x3_wmma<<<grid, 128, 0, stream>>>(in, w, b, out, 2, Ci, Co, H, W, relu,
                                               pow2_shift_ge32(Ci));
    };
    auto pool = [&](const float* in, float* out, int BC, int H, int W) {
        int tot = BC * (H / 2) * (W / 2);
        maxpool2_kernel<<<ceil_div_i(tot, 256), 256, 0, stream>>>(in, out, BC, H, W);
    };
    auto upconv = [&](const float* in, const float* tw, const float* tb, float* cat,
                      int Ci, int Co, int H, int W, int Ctot) {
        for (int oy = 0; oy < 2; ++oy)
            for (int ox = 0; ox < 2; ++ox) {
                dim3 grid(ceil_div_i(2 * H * W, 128), ceil_div_i(Co, 16));
                gemm_pix_wmma<<<grid, 128, 0, stream>>>(in, tw, tb, cat, 2, Ci, Co, H, W,
                                                        4, oy * 2 + ox, 2, oy, 2, ox,
                                                        Ctot, 0, 0, pow2_shift_ge32(Ci));
            }
    };
    auto cpycat = [&](const float* src, float* dst, int C, int HW, int Ctot, int cb) {
        int tot = 2 * C * HW;
        copy_into_cat<<<ceil_div_i(tot, 256), 256, 0, stream>>>(src, dst, 2, C, HW, Ctot, cb);
    };

    // UNet param layout within [pb, pb+36): bott w1,b1,w2,b2 | enc{0,1,2} w1,b1,w2,b2
    //   | out w,b | up{0,1,2}: dc w1,b1,w2,b2, tb, tw
    auto run_unet = [&](int pb, float* featOut) {
        conv3(image, Pp(pb + 4),  Pp(pb + 5),  tmp, 3,   16, 256, 256, 1);
        conv3(tmp,   Pp(pb + 6),  Pp(pb + 7),  e1,  16,  16, 256, 256, 1);
        pool(e1, e1p, 2 * 16, 256, 256);
        conv3(e1p,   Pp(pb + 8),  Pp(pb + 9),  tmp, 16,  32, 128, 128, 1);
        conv3(tmp,   Pp(pb + 10), Pp(pb + 11), e2,  32,  32, 128, 128, 1);
        pool(e2, e2p, 2 * 32, 128, 128);
        conv3(e2p,   Pp(pb + 12), Pp(pb + 13), tmp, 32,  64, 64, 64, 1);
        conv3(tmp,   Pp(pb + 14), Pp(pb + 15), e3,  64,  64, 64, 64, 1);
        pool(e3, e3p, 2 * 64, 64, 64);
        conv3(e3p,   Pp(pb + 0),  Pp(pb + 1),  bA,  64,  128, 32, 32, 1);
        conv3(bA,    Pp(pb + 2),  Pp(pb + 3),  bB,  128, 128, 32, 32, 1);
        // up0: 128 -> 64, 32^2 -> 64^2
        upconv(bB, Pp(pb + 23), Pp(pb + 22), cat3, 128, 64, 32, 32, 128);
        cpycat(e3, cat3, 64, 64 * 64, 128, 64);
        conv3(cat3,  Pp(pb + 18), Pp(pb + 19), tmp, 128, 64, 64, 64, 1);
        conv3(tmp,   Pp(pb + 20), Pp(pb + 21), d3,  64,  64, 64, 64, 1);
        // up1: 64 -> 32, 64^2 -> 128^2
        upconv(d3, Pp(pb + 29), Pp(pb + 28), cat2, 64, 32, 64, 64, 64);
        cpycat(e2, cat2, 32, 128 * 128, 64, 32);
        conv3(cat2,  Pp(pb + 24), Pp(pb + 25), tmp, 64,  32, 128, 128, 1);
        conv3(tmp,   Pp(pb + 26), Pp(pb + 27), d2,  32,  32, 128, 128, 1);
        // up2: 32 -> 16, 128^2 -> 256^2
        upconv(d2, Pp(pb + 35), Pp(pb + 34), cat1, 32, 16, 128, 128, 32);
        cpycat(e1, cat1, 16, 256 * 256, 32, 16);
        conv3(cat1,  Pp(pb + 30), Pp(pb + 31), tmp, 32,  16, 256, 256, 1);
        conv3(tmp,   Pp(pb + 32), Pp(pb + 33), d1,  16,  16, 256, 256, 1);
        // out 1x1: 16 -> 128
        dim3 grid(ceil_div_i(2 * 256 * 256, 128), ceil_div_i(128, 16));
        gemm_pix_wmma<<<grid, 128, 0, stream>>>(d1, Pp(pb + 16), Pp(pb + 17), featOut,
                                                2, 16, 128, 256, 256,
                                                1, 0, 1, 0, 1, 0, 128, 0, 0,
                                                pow2_shift_ge32(16));
    };

    auto run_mlp = [&](int pbase, float* dstBase, int cout) {
        const int dins[5]  = {131, 131, 64, 64, 32};
        const int douts[5] = {131, 64, 64, 32, cout};
        const float* cur = mlpA;
        float* pong[2] = {mlpB, mlpA};
        int pp = 0;
        for (int i = 0; i < 5; ++i) {
            int K = dins[i], Nout = douts[i];
            float* dst   = (i == 4) ? dstBase : pong[pp];
            int    ldo   = (i == 4) ? cout : MLP_LD;
            int    act   = (i == 4) ? 1 : 0;          // relu hidden, sigmoid last
            int    padTo = (i == 4) ? 0 : MLP_LD;     // zero-fill padding cols
            int    ncols = (i == 4) ? Nout : MLP_LD;
            dim3 grid(NPTS / 128, ceil_div_i(ncols, 16));
            mlp_wmma<<<grid, 128, 0, stream>>>(cur, Pp(pbase + 2 * i), Pp(pbase + 2 * i + 1),
                                               dst, NPTS, K, Nout, ldo, act, padTo);
            cur = dst;
            pp ^= 1;
        }
    };

    // coordinate min/max (shared by both branches — coo is identical)
    minmax_init<<<1, 32, 0, stream>>>(stats);
    minmax_coo<<<ceil_div_i(NPTS, 256), 256, 0, stream>>>(coo, NPTS, stats);

    // ---- geometry branch -> occ ----
    run_unet(2, feat);
    interp_gather<<<ceil_div_i(NPTS * MLP_LD, 256), 256, 0, stream>>>(
        feat, coo, stats, mlpA, NPTS, 32768, 128, 256, 256, MLP_LD);
    run_mlp(74, outp, 1);                 // occ: (B,N,1) -> d_out[0 .. 65535]

    // ---- rendering branch -> rgb ----
    run_unet(38, feat);
    interp_gather<<<ceil_div_i(NPTS * MLP_LD, 256), 256, 0, stream>>>(
        feat, coo, stats, mlpA, NPTS, 32768, 128, 256, 256, MLP_LD);
    run_mlp(84, outp + NPTS, 3);          // rgb: (B,N,3) -> d_out[65536 ..]
}
